// CausalAttNet_9131100471736
// MI455X (gfx1250) — matrix operand order repaired and checked by reading, not compiled, not run on hardware
//
#include <hip/hip_runtime.h>
#include <hip/hip_bf16.h>

// ---------------------------------------------------------------------------
// Problem constants (from the reference)
// ---------------------------------------------------------------------------
constexpr int       Bg      = 128;          // graphs
constexpr int       NPG     = 256;          // nodes per graph
constexpr int       EPG     = 4096;         // edges per graph
constexpr int       Nn      = Bg * NPG;     // 32768 nodes
constexpr long long Ee      = (long long)Bg * EPG; // 524288 edges
constexpr int       IN_DIM  = 128;
constexpr int       MED     = 32;
constexpr int       KCHUNK  = 4;
constexpr int       KCAUSAL = EPG / 2;      // 2048
constexpr long long NC      = (long long)Bg * KCAUSAL; // 262144

// Output layout (flat f32, reference return order)
constexpr long long SZ_X      = (long long)Nn * IN_DIM;
constexpr long long OFF_CX    = 0;
constexpr long long OFF_CEI   = OFF_CX   + SZ_X;
constexpr long long OFF_CATTR = OFF_CEI  + 2 * NC;
constexpr long long OFF_CW    = OFF_CATTR + NC;
constexpr long long OFF_CB    = OFF_CW   + NC;
constexpr long long OFF_FX    = OFF_CB   + Nn;
constexpr long long OFF_FEI   = OFF_FX   + SZ_X;
constexpr long long OFF_FATTR = OFF_FEI  + 2 * NC;
constexpr long long OFF_FW    = OFF_FATTR + NC;
constexpr long long OFF_FB    = OFF_FW   + NC;
constexpr long long OFF_SCORE = OFF_FB   + Nn;

// ---------------------------------------------------------------------------
// WMMA plumbing (wave32, 16x16x32 f16 -> f32)
// ---------------------------------------------------------------------------
typedef __attribute__((ext_vector_type(16))) _Float16 v16h;
typedef __attribute__((ext_vector_type(8)))  float    v8f;

__device__ __forceinline__ v8f wmma_f16(v16h a, v16h b, v8f c) {
  return __builtin_amdgcn_wmma_f32_16x16x32_f16(
      /*neg_a=*/false, a, /*neg_b=*/false, b,
      /*c_mod=*/(short)0, c, /*reuse_a=*/false, /*reuse_b=*/false);
}

// ISA A-fragment K pattern for a 16x32 f16 tile: lanes 0-15 need K={0..7,16..23},
// lanes 16-31 need K={8..15,24..31}. Storing features with this involutive
// permutation makes each lane's 16 halves CONTIGUOUS (one 32-byte load):
//   kperm: 0..7 -> 0..7 ; 8..15 -> 16..23 ; 16..23 -> 8..15 ; 24..31 -> 24..31
__device__ __forceinline__ int kperm(int f) {
  return (f & 7) | ((f & 8) << 1) | ((f & 16) >> 1);
}
// C fragment element r maps to (M = r + (lane<16?0:8), N = lane&15).

__device__ __forceinline__ float frelu(float v) { return v > 0.0f ? v : 0.0f; }

// ---------------------------------------------------------------------------
// Prep A: generic f32 copy (agg init)
// ---------------------------------------------------------------------------
__global__ void k_copyf(const float* __restrict__ src, float* __restrict__ dst,
                        long long n) {
  long long i = blockIdx.x * (long long)blockDim.x + threadIdx.x;
  const long long stride = gridDim.x * (long long)blockDim.x;
  for (; i < n; i += stride) dst[i] = src[i];
}

// ---------------------------------------------------------------------------
// Prep B: weight -> f16 fragment-order buffer.
// dst[((c*T + t)*32 + lane)*16 + j] = src[(c*32 + kperm(kofs(lane) + j))*ld + t*16 + (lane&15)]
// so a B fragment (32x16 tile, K chunk c, N tile t) is 32 contiguous bytes/lane.
// ---------------------------------------------------------------------------
__global__ void k_prep_bfrag(const float* __restrict__ src, _Float16* __restrict__ dst,
                             int C, int T, int ld) {
  const int i = blockIdx.x * blockDim.x + threadIdx.x;
  const int total = C * T * 32 * 16;
  if (i >= total) return;
  const int j    = i & 15;
  const int lane = (i >> 4) & 31;
  const int t    = (i >> 9) % T;
  const int c    = (i >> 9) / T;
  const int kofs = (lane < 16) ? 0 : 16;
  const int n    = lane & 15;
  dst[i] = (_Float16)src[(long long)(c * 32 + kperm(kofs + j)) * ld + t * 16 + n];
}

// ---------------------------------------------------------------------------
// Prep C: agg1 (N x 128 f32) -> permuted f16 (per 32-chunk kperm layout)
// ---------------------------------------------------------------------------
__global__ void k_cvt_agg1(const float* __restrict__ agg1, _Float16* __restrict__ agg1h) {
  long long i = blockIdx.x * (long long)blockDim.x + threadIdx.x;
  const long long stride = gridDim.x * (long long)blockDim.x;
  const long long total = (long long)Nn * IN_DIM;
  for (; i < total; i += stride) {
    const long long node = i >> 7;
    const int f  = (int)(i & 127);
    const int kc = f >> 5, fi = f & 31;
    agg1h[i] = (_Float16)agg1[node * IN_DIM + kc * 32 + kperm(fi)];  // kperm is an involution
  }
}

// ---------------------------------------------------------------------------
// GIN aggregation scatter: agg[dst] += feat[src] (agg pre-initialized = feat)
// agg buffers fit in the 192 MB L2, so these 67M float atomics stay on-chip.
// ---------------------------------------------------------------------------
__global__ void k_scatter(const float* __restrict__ feat,
                          const long long* __restrict__ esrc,
                          const long long* __restrict__ edst,
                          float* __restrict__ agg, int F) {
  long long i = blockIdx.x * (long long)blockDim.x + threadIdx.x;
  const long long stride = gridDim.x * (long long)blockDim.x;
  const long long total = Ee * F;
  for (; i < total; i += stride) {
    const long long e = i / F;
    const int       f = (int)(i - e * F);
    const int s = (int)esrc[e];
    const int d = (int)edst[e];
    atomicAdd(&agg[(long long)d * F + f], feat[(long long)s * F + f]);
  }
}

// ---------------------------------------------------------------------------
// conv1 = relu( relu(h @ W1 + b1) @ W2 + b2 ); seeds agg2 for the 2nd agg.
// One wave per 16-node tile; 8 waves/block. All WMMA operands are single
// 32-byte contiguous f16 loads (global b128x2 or ds b128x2).
// ---------------------------------------------------------------------------
__global__ void k_conv1(const _Float16* __restrict__ agg1h,
                        const _Float16* __restrict__ wf1, const float* __restrict__ b1,
                        const _Float16* __restrict__ wf2, const float* __restrict__ b2,
                        float* __restrict__ h1, float* __restrict__ agg2) {
  __shared__ alignas(32) _Float16 midh[8][16 * MED];
  const int wv   = threadIdx.x >> 5;
  const int lane = threadIdx.x & 31;
  const int tile = blockIdx.x * 8 + wv;
  const long long base = (long long)tile * 16;

  const int row  = lane & 15;
  const int aoff = (lane < 16) ? 0 : 16;

  // Stage 1: (16 x 128) @ (128 x 32), K in 4 chunks of 32
  v8f acc0 = {}, acc1 = {};
#pragma unroll
  for (int kc = 0; kc < 4; ++kc) {
    const v16h a   = *(const v16h*)(agg1h + (base + row) * IN_DIM + kc * 32 + aoff);
    const v16h bf0 = *(const v16h*)(wf1 + ((kc * 2 + 0) * 32 + lane) * 16);
    const v16h bf1 = *(const v16h*)(wf1 + ((kc * 2 + 1) * 32 + lane) * 16);
    acc0 = wmma_f16(a, bf0, acc0);
    acc1 = wmma_f16(a, bf1, acc1);
  }
  const int n    = lane & 15;
  const int moff = (lane < 16) ? 0 : 8;
  _Float16* m = &midh[wv][0];
#pragma unroll
  for (int r = 0; r < 8; ++r) {   // store C tile as permuted f16 -> stage-2 A frags
    m[(moff + r) * MED + kperm(n)]      = (_Float16)frelu(acc0[r] + b1[n]);
    m[(moff + r) * MED + kperm(16 + n)] = (_Float16)frelu(acc1[r] + b1[16 + n]);
  }
  __syncthreads();

  // Stage 2: (16 x 32) @ (32 x 32), single K chunk
  const v16h a2  = *(const v16h*)(m + row * MED + aoff);
  const v16h cf0 = *(const v16h*)(wf2 + (0 * 32 + lane) * 16);
  const v16h cf1 = *(const v16h*)(wf2 + (1 * 32 + lane) * 16);
  v8f o0 = {}, o1 = {};
  o0 = wmma_f16(a2, cf0, o0);
  o1 = wmma_f16(a2, cf1, o1);
#pragma unroll
  for (int r = 0; r < 8; ++r) {
    const float v0 = frelu(o0[r] + b2[n]);
    const float v1 = frelu(o1[r] + b2[16 + n]);
    const long long rr = (base + moff + r) * MED;
    h1[rr + n]        = v0;  h1[rr + 16 + n]   = v1;
    agg2[rr + n]      = v0;  agg2[rr + 16 + n] = v1;  // seed for 2nd agg
  }
}

// ---------------------------------------------------------------------------
// conv2 — per (node, chunk) 8x8 matmuls (tiny; pure VALU). Emits h2 directly
// as permuted f16 so the edge-MLP A fragments are contiguous 32-byte gathers.
// ---------------------------------------------------------------------------
__global__ void k_conv2(const float* __restrict__ agg2,
                        const float* __restrict__ w1, const float* __restrict__ b1,
                        const float* __restrict__ w2, const float* __restrict__ b2,
                        _Float16* __restrict__ h2h) {
  const int t = blockIdx.x * blockDim.x + threadIdx.x;
  if (t >= Nn * KCHUNK) return;
  const int node = t >> 2, k = t & 3;
  const float* a = agg2 + (long long)node * MED + k * 8;
  float z[8];
#pragma unroll
  for (int e = 0; e < 8; ++e) {
    float s = b1[k * 8 + e];
#pragma unroll
    for (int d = 0; d < 8; ++d) s += a[d] * w1[k * 64 + d * 8 + e];
    z[e] = frelu(s);
  }
#pragma unroll
  for (int e = 0; e < 8; ++e) {
    float s = b2[k * 8 + e];
#pragma unroll
    for (int d = 0; d < 8; ++d) s += z[d] * w2[k * 64 + d * 8 + e];
    h2h[(long long)node * MED + kperm(k * 8 + e)] = (_Float16)s;
  }
}

// ---------------------------------------------------------------------------
// Fused edge MLP + score. One wave per 16-edge tile, 8 waves/block.
// hid (E x 128) is never materialized (would be 256 MB > L2); it lives in
// accumulator VGPRs. Weights are LDS-staged once per block (16 KB f16 frags).
// ---------------------------------------------------------------------------
__global__ void k_edge_mlp(const _Float16* __restrict__ h2h,
                           const long long* __restrict__ esrc,
                           const long long* __restrict__ edst,
                           const _Float16* __restrict__ wf,   // [2][8][32][16] f16 frags
                           const float* __restrict__ b1,
                           const float* __restrict__ w2, const float* __restrict__ b2,
                           float* __restrict__ score) {
  __shared__ alignas(32) _Float16 wlds[2 * 8 * 32 * 16];   // 16 KB

  // Cooperative stage of all weight fragments into LDS (b128 copies)
  {
    const uint4* s4 = (const uint4*)wf;
    uint4*       d4 = (uint4*)wlds;
    for (int i = threadIdx.x; i < (2 * 8 * 32 * 16) / 8; i += blockDim.x) d4[i] = s4[i];
  }
  __syncthreads();

  const int wave = (blockIdx.x * blockDim.x + threadIdx.x) >> 5;
  const int lane = threadIdx.x & 31;
  const long long ebase = (long long)wave * 16;

  __builtin_prefetch(h2h, 0, 0);                 // global_prefetch_b8

  // Gathered A fragments: chunk 0 <- h2[src], chunk 1 <- h2[dst]
  const int erow = lane & 15;
  const int ns = (int)esrc[ebase + erow];
  const int nd = (int)edst[ebase + erow];
  const int aoff = (lane < 16) ? 0 : 16;
  const v16h a0 = *(const v16h*)(h2h + (long long)ns * MED + aoff);
  const v16h a1 = *(const v16h*)(h2h + (long long)nd * MED + aoff);

  float part[8];
#pragma unroll
  for (int r = 0; r < 8; ++r) part[r] = 0.0f;
  const float b2v = b2[0];

#pragma unroll
  for (int t = 0; t < 8; ++t) {                 // 8 column tiles of N=128
    const v16h bf0 = *(const v16h*)(wlds + ((0 * 8 + t) * 32 + lane) * 16);
    const v16h bf1 = *(const v16h*)(wlds + ((1 * 8 + t) * 32 + lane) * 16);
    v8f acc = {};
    acc = wmma_f16(a0, bf0, acc);
    acc = wmma_f16(a1, bf1, acc);
    const int n = t * 16 + (lane & 15);
    const float bias = b1[n], wv = w2[n];
#pragma unroll
    for (int r = 0; r < 8; ++r) part[r] += frelu(acc[r] + bias) * wv;
  }

  // Reduce across the 16 lanes sharing each row group
#pragma unroll
  for (int m = 1; m < 16; m <<= 1) {
#pragma unroll
    for (int r = 0; r < 8; ++r) part[r] += __shfl_xor(part[r], m, 16);
  }
  if ((lane & 15) == 0) {
    const int moff = (lane < 16) ? 0 : 8;
#pragma unroll
    for (int r = 0; r < 8; ++r) score[ebase + moff + r] = part[r] + b2v;
  }
}

// ---------------------------------------------------------------------------
// Per-graph descending bitonic argsort of 4096 scores in LDS (32 KB/block)
// ---------------------------------------------------------------------------
__global__ void k_sort(const float* __restrict__ score, int* __restrict__ order) {
  __shared__ float skey[EPG];
  __shared__ int   sidx[EPG];
  const int row = blockIdx.x;
  const float* s = score + (long long)row * EPG;
  for (int i = threadIdx.x; i < EPG; i += blockDim.x) { skey[i] = s[i]; sidx[i] = i; }
  __syncthreads();
  for (int size = 2; size <= EPG; size <<= 1) {
    for (int stride = size >> 1; stride > 0; stride >>= 1) {
      for (int i = threadIdx.x; i < EPG / 2; i += blockDim.x) {
        const int pos = 2 * i - (i & (stride - 1));
        const int a = pos, b = pos + stride;
        const bool desc = ((pos & size) == 0);
        const bool doswap = desc ? (skey[a] < skey[b]) : (skey[a] > skey[b]);
        if (doswap) {
          const float tk = skey[a]; skey[a] = skey[b]; skey[b] = tk;
          const int   ti = sidx[a]; sidx[a] = sidx[b]; sidx[b] = ti;
        }
      }
      __syncthreads();
    }
  }
  for (int i = threadIdx.x; i < EPG; i += blockDim.x)
    order[(long long)row * EPG + i] = sidx[i];
}

// ---------------------------------------------------------------------------
// x / batch / edge_score passthrough outputs
// ---------------------------------------------------------------------------
__global__ void k_write_misc(const float* __restrict__ x,
                             const long long* __restrict__ batch,
                             const float* __restrict__ score,
                             float* __restrict__ out) {
  long long i = blockIdx.x * (long long)blockDim.x + threadIdx.x;
  const long long stride = gridDim.x * (long long)blockDim.x;
  for (long long t = i; t < SZ_X; t += stride) {
    const float v = x[t];
    out[OFF_CX + t] = v;
    out[OFF_FX + t] = v;
    if (t < Nn) {
      const float bv = (float)batch[t];
      out[OFF_CB + t] = bv;
      out[OFF_FB + t] = bv;
    }
    if (t < Ee) out[OFF_SCORE + t] = score[t];
  }
}

// ---------------------------------------------------------------------------
// Split edges into causal (top-K) / conf via the sorted order
// ---------------------------------------------------------------------------
__global__ void k_write_edges(const long long* __restrict__ esrc,
                              const long long* __restrict__ edst,
                              const long long* __restrict__ eattr,
                              const float* __restrict__ score,
                              const int* __restrict__ order,
                              float* __restrict__ out) {
  const long long t = blockIdx.x * (long long)blockDim.x + threadIdx.x;
  if (t >= Ee) return;
  const int b    = (int)(t >> 12);          // / EPG
  const int rank = (int)(t & (EPG - 1));
  const long long eid = (long long)b * EPG + order[t];
  const float s = score[eid];
  if (rank < KCAUSAL) {
    const long long p = (long long)b * KCAUSAL + rank;
    out[OFF_CEI + p]      = (float)esrc[eid];
    out[OFF_CEI + NC + p] = (float)edst[eid];
    out[OFF_CATTR + p]    = (float)eattr[eid];
    out[OFF_CW + p]       = s;
  } else {
    const long long p = (long long)b * KCAUSAL + (rank - KCAUSAL);
    out[OFF_FEI + p]      = (float)esrc[eid];
    out[OFF_FEI + NC + p] = (float)edst[eid];
    out[OFF_FATTR + p]    = (float)eattr[eid];
    out[OFF_FW + p]       = -s;
  }
}

// ---------------------------------------------------------------------------
// Host launch
// ---------------------------------------------------------------------------
extern "C" void kernel_launch(void* const* d_in, const int* in_sizes, int n_in,
                              void* d_out, int out_size, void* d_ws, size_t ws_size,
                              hipStream_t stream) {
  const float*     x     = (const float*)d_in[0];
  const long long* ei    = (const long long*)d_in[1];
  const long long* batch = (const long long*)d_in[2];
  const long long* eattr = (const long long*)d_in[3];
  const float* c1w1 = (const float*)d_in[4];
  const float* c1b1 = (const float*)d_in[5];
  const float* c1w2 = (const float*)d_in[6];
  const float* c1b2 = (const float*)d_in[7];
  const float* c2w1 = (const float*)d_in[8];
  const float* c2b1 = (const float*)d_in[9];
  const float* c2w2 = (const float*)d_in[10];
  const float* c2b2 = (const float*)d_in[11];
  const float* mw1  = (const float*)d_in[12];
  const float* mb1  = (const float*)d_in[13];
  const float* mw2  = (const float*)d_in[14];
  const float* mb2  = (const float*)d_in[15];

  const long long* esrc = ei;
  const long long* edst = ei + Ee;

  // Workspace carve-up (bytes); everything fits in the 192 MB L2 (~40 MB).
  char* ws = (char*)d_ws;
  float*     agg1   = (float*)ws;                 ws += (long long)Nn * IN_DIM * 4; // 16 MB
  _Float16*  agg1h  = (_Float16*)ws;              ws += (long long)Nn * IN_DIM * 2; //  8 MB
  float*     h1     = (float*)ws;                 ws += (long long)Nn * MED * 4;    //  4 MB
  float*     agg2   = (float*)ws;                 ws += (long long)Nn * MED * 4;    //  4 MB
  _Float16*  h2h    = (_Float16*)ws;              ws += (long long)Nn * MED * 2;    //  2 MB
  float*     score  = (float*)ws;                 ws += Ee * 4;                     //  2 MB
  int*       order  = (int*)ws;                   ws += Ee * 4;                     //  2 MB
  _Float16*  wfmlp  = (_Float16*)ws;              ws += 2 * 8 * 32 * 16 * 2;        // 16 KB
  _Float16*  wfc11  = (_Float16*)ws;              ws += 4 * 2 * 32 * 16 * 2;        //  8 KB
  _Float16*  wfc12  = (_Float16*)ws;              ws += 1 * 2 * 32 * 16 * 2;        //  2 KB

  float* out = (float*)d_out;

  // 0) one-time weight fragment prep (f16, fragment order)
  k_prep_bfrag<<<(2 * 8 * 512 + 255) / 256, 256, 0, stream>>>(mw1,  wfmlp, 2, 8, IN_DIM);
  k_prep_bfrag<<<(4 * 2 * 512 + 255) / 256, 256, 0, stream>>>(c1w1, wfc11, 4, 2, MED);
  k_prep_bfrag<<<(1 * 2 * 512 + 255) / 256, 256, 0, stream>>>(c1w2, wfc12, 1, 2, MED);

  // 1) agg1 = x ; agg1 += scatter(x) ; convert to permuted f16
  k_copyf<<<4096, 256, 0, stream>>>(x, agg1, (long long)Nn * IN_DIM);
  k_scatter<<<8192, 256, 0, stream>>>(x, esrc, edst, agg1, IN_DIM);
  k_cvt_agg1<<<4096, 256, 0, stream>>>(agg1, agg1h);

  // 2) conv1 (WMMA): 2048 node tiles, 8 waves/block
  k_conv1<<<Nn / 16 / 8, 256, 0, stream>>>(agg1h, wfc11, c1b1, wfc12, c1b2, h1, agg2);

  // 3) agg2 (seeded with h1 inside k_conv1) += scatter(h1)
  k_scatter<<<4096, 256, 0, stream>>>(h1, esrc, edst, agg2, MED);

  // 4) conv2 (tiny per-chunk 8x8) -> permuted f16 h2
  k_conv2<<<(Nn * KCHUNK + 255) / 256, 256, 0, stream>>>(agg2, c2w1, c2b1, c2w2, c2b2, h2h);

  // 5) fused edge MLP + score (WMMA): 32768 edge tiles, 8 waves/block
  k_edge_mlp<<<(int)(Ee / 16 / 8), 256, 0, stream>>>(h2h, esrc, edst, wfmlp, mb1, mw2, mb2, score);

  // 6) per-graph argsort
  k_sort<<<Bg, 512, 0, stream>>>(score, order);

  // 7) outputs
  k_write_misc<<<4096, 256, 0, stream>>>(x, batch, score, out);
  k_write_edges<<<(int)((Ee + 255) / 256), 256, 0, stream>>>(esrc, edst, eattr, score, order, out);
}
